// GlobalDensityApprox_40295383171135
// MI455X (gfx1250) — matrix operand order repaired and checked by reading, not compiled, not run on hardware
//
#include <hip/hip_runtime.h>

// ---------------- problem constants ----------------
#define BB 4
#define NN 4096
#define PP 512
#define EE 256
#define WW 1024
#define NBASIS 4
#define NHEADS_ 8
#define HD_ 32
#define NBLOCKS_ 4

#define LDA_C_ (-0.73855876638202234f)
#define LN2_   (0.69314718055994531f)

// ---------------- workspace layout (float offsets) ----------------
#define WS_WRHO 0
#define WS_STAT (WS_WRHO + BB*NN)
#define ST_SUMW (WS_STAT + 0)     /* [B]   */
#define ST_ELDA (WS_STAT + 4)     /* [B]   */
#define ST_MSUM (WS_STAT + 8)     /* [B*3] */
#define ST_COV  (WS_STAT + 20)    /* [B*6] */
#define ST_S2   (WS_STAT + 48)    /* [B*3] */
#define ST_R    (WS_STAT + 64)    /* [B*9] */
#define ST_DST  (WS_STAT + 104)   /* [B*2] */
#define WS_ANCH (WS_STAT + 256)               /* B*P*3   */
#define WS_CR   (WS_ANCH + BB*PP*3)           /* B*N*3   */
#define WS_PHI  (WS_CR   + BB*NN*3)           /* B*P*NB  */
#define WS_DIST (WS_PHI  + BB*PP*NBASIS)      /* B*P*P   */
#define WS_X    (WS_DIST + BB*PP*PP)          /* B*P*E   */
#define WS_Q    (WS_X    + BB*PP*EE)
#define WS_K    (WS_Q    + BB*PP*EE)
#define WS_V    (WS_K    + BB*PP*EE)
#define WS_O    (WS_V    + BB*PP*EE)
#define WS_O2   (WS_O    + BB*PP*EE)
#define WS_H    (WS_O2   + BB*PP*EE)          /* B*P*W   */

// ---------------- WMMA types ----------------
typedef __attribute__((ext_vector_type(16))) __bf16 bf16x16;
typedef __attribute__((ext_vector_type(8)))  float  floatx8;

union FragAB {
    bf16x16 v;
    __bf16  h[16];
    uint4   q[2];
};

__device__ __forceinline__ floatx8 wmma_bf16(bf16x16 a, bf16x16 b, floatx8 c) {
    return __builtin_amdgcn_wmma_f32_16x16x32_bf16(false, a, false, b, (short)0, c, false, false);
}

__device__ __forceinline__ float fast_rcp(float x) {
    return __builtin_amdgcn_rcpf(x);   // v_rcp_f32 (1 ulp) — fine for bf16-fed activations
}

__device__ __forceinline__ float fast_silu(float v) {
    return v * fast_rcp(1.f + __expf(-v));
}

__device__ __forceinline__ float block_reduce(float v, float* red) {
    int t = threadIdx.x;
    red[t] = v;
    __syncthreads();
    for (int s = 128; s > 0; s >>= 1) {
        if (t < s) red[t] += red[t + s];
        __syncthreads();
    }
    float r = red[0];
    __syncthreads();
    return r;
}

// ---------------- stage 1: weighted moments + e_lda + wrho ----------------
__global__ __launch_bounds__(256) void k_stats1(const float* __restrict__ rho,
                                                const float* __restrict__ coords,
                                                const float* __restrict__ weights,
                                                float* __restrict__ ws) {
    __shared__ float red[256];
    int b = blockIdx.x, t = threadIdx.x;
    float s_w = 0.f, s_e = 0.f, m0 = 0.f, m1 = 0.f, m2 = 0.f;
    for (int n = t; n < NN; n += 256) {
        float r  = rho[b*NN + n];
        float wr = weights[b*NN + n] * r;
        ws[WS_WRHO + b*NN + n] = wr;
        s_w += wr;
        s_e += wr * (LDA_C_ * cbrtf(fmaxf(r, 1e-7f)));
        const float* c = coords + ((size_t)(b*NN + n)) * 3;
        m0 += wr * c[0]; m1 += wr * c[1]; m2 += wr * c[2];
    }
    float tw = block_reduce(s_w, red);
    float te = block_reduce(s_e, red);
    float t0 = block_reduce(m0,  red);
    float t1 = block_reduce(m1,  red);
    float t2 = block_reduce(m2,  red);
    if (t == 0) {
        ws[ST_SUMW + b] = tw;
        ws[ST_ELDA + b] = te;
        ws[ST_MSUM + b*3 + 0] = t0;
        ws[ST_MSUM + b*3 + 1] = t1;
        ws[ST_MSUM + b*3 + 2] = t2;
    }
}

// ---------------- stage 2: weighted covariance ----------------
__global__ __launch_bounds__(256) void k_cov(const float* __restrict__ coords,
                                             float* __restrict__ ws) {
    __shared__ float red[256];
    int b = blockIdx.x, t = threadIdx.x;
    float sumw = ws[ST_SUMW + b];
    float m0 = ws[ST_MSUM + b*3 + 0] / sumw;
    float m1 = ws[ST_MSUM + b*3 + 1] / sumw;
    float m2 = ws[ST_MSUM + b*3 + 2] / sumw;
    float c00=0.f,c01=0.f,c02=0.f,c11=0.f,c12=0.f,c22=0.f;
    for (int n = t; n < NN; n += 256) {
        float w = ws[WS_WRHO + b*NN + n];
        const float* cp = coords + ((size_t)(b*NN + n)) * 3;
        float x0 = cp[0]-m0, x1 = cp[1]-m1, x2 = cp[2]-m2;
        c00 += w*x0*x0; c01 += w*x0*x1; c02 += w*x0*x2;
        c11 += w*x1*x1; c12 += w*x1*x2; c22 += w*x2*x2;
    }
    float s0 = block_reduce(c00, red);
    float s1 = block_reduce(c01, red);
    float s2 = block_reduce(c02, red);
    float s3 = block_reduce(c11, red);
    float s4 = block_reduce(c12, red);
    float s5 = block_reduce(c22, red);
    if (t == 0) {
        float inv = 1.0f / sumw;
        ws[ST_COV + b*6 + 0] = s0*inv; ws[ST_COV + b*6 + 1] = s1*inv;
        ws[ST_COV + b*6 + 2] = s2*inv; ws[ST_COV + b*6 + 3] = s3*inv;
        ws[ST_COV + b*6 + 4] = s4*inv; ws[ST_COV + b*6 + 5] = s5*inv;
    }
}

// ---------------- stage 3: 3x3 symmetric eigensolve (Jacobi, ascending) ----------------
__global__ void k_eigen(float* __restrict__ ws) {
    int b = threadIdx.x;
    if (b >= BB) return;
    const float* cv = ws + ST_COV + b*6;
    float a[3][3];
    a[0][0]=cv[0]; a[0][1]=cv[1]; a[0][2]=cv[2];
    a[1][0]=cv[1]; a[1][1]=cv[3]; a[1][2]=cv[4];
    a[2][0]=cv[2]; a[2][1]=cv[4]; a[2][2]=cv[5];
    float V[3][3] = {{1.f,0.f,0.f},{0.f,1.f,0.f},{0.f,0.f,1.f}};
    for (int sweep = 0; sweep < 16; ++sweep) {
        const int prs[3][2] = {{0,1},{0,2},{1,2}};
        for (int it = 0; it < 3; ++it) {
            int p = prs[it][0], q = prs[it][1];
            float apq = a[p][q];
            if (fabsf(apq) < 1e-20f) continue;
            float theta = (a[q][q] - a[p][p]) / (2.0f * apq);
            float tt = ((theta >= 0.f) ? 1.f : -1.f) / (fabsf(theta) + sqrtf(theta*theta + 1.f));
            float c = rsqrtf(tt*tt + 1.f), s = tt*c;
            for (int k = 0; k < 3; ++k) { float akp=a[k][p], akq=a[k][q]; a[k][p]=c*akp - s*akq; a[k][q]=s*akp + c*akq; }
            for (int k = 0; k < 3; ++k) { float apk=a[p][k], aqk=a[q][k]; a[p][k]=c*apk - s*aqk; a[q][k]=s*apk + c*aqk; }
            for (int k = 0; k < 3; ++k) { float vkp=V[k][p], vkq=V[k][q]; V[k][p]=c*vkp - s*vkq; V[k][q]=s*vkp + c*vkq; }
        }
    }
    float d[3] = {a[0][0], a[1][1], a[2][2]};
    int idx[3] = {0,1,2};
    for (int i = 0; i < 2; ++i)
        for (int j = 0; j < 2-i; ++j)
            if (d[idx[j]] > d[idx[j+1]]) { int tm = idx[j]; idx[j] = idx[j+1]; idx[j+1] = tm; }
    for (int i = 0; i < 3; ++i) {
        ws[ST_S2 + b*3 + i] = d[idx[i]];
        for (int r = 0; r < 3; ++r) ws[ST_R + b*9 + r*3 + i] = V[r][idx[i]];
    }
}

// ---------------- anchors & rotated coords ----------------
__global__ __launch_bounds__(256) void k_anchors(const float* __restrict__ grid,
                                                 float* __restrict__ ws) {
    int idx = blockIdx.x * 256 + threadIdx.x;
    if (idx >= BB*PP) return;
    int b = idx / PP, p = idx % PP;
    for (int i = 0; i < 3; ++i)
        ws[WS_ANCH + (size_t)idx*3 + i] = sqrtf(ws[ST_S2 + b*3 + i] + 1e-5f) * grid[p*3 + i];
}

__global__ __launch_bounds__(256) void k_coordsr(const float* __restrict__ coords,
                                                 float* __restrict__ ws) {
    int idx = blockIdx.x * 256 + threadIdx.x;
    if (idx >= BB*NN) return;
    int b = idx / NN;
    float sumw = ws[ST_SUMW + b];
    float xc[3];
    for (int i = 0; i < 3; ++i)
        xc[i] = coords[(size_t)idx*3 + i] - ws[ST_MSUM + b*3 + i] / sumw;
    const float* R = ws + ST_R + b*9;
    for (int j = 0; j < 3; ++j)
        ws[WS_CR + (size_t)idx*3 + j] = xc[0]*R[j*3+0] + xc[1]*R[j*3+1] + xc[2]*R[j*3+2];
}

// ---------------- Gaussian pooling -> log phi ----------------
__global__ __launch_bounds__(256) void k_gauss(float* __restrict__ ws) {
    __shared__ float red[256];
    int bp = blockIdx.x, t = threadIdx.x;
    int b = bp >> 9;  // P = 512
    const float* an = ws + WS_ANCH + (size_t)bp*3;
    float a0 = an[0], a1 = an[1], a2 = an[2];
    const float* cr = ws + WS_CR + (size_t)b*NN*3;
    const float* wr = ws + WS_WRHO + (size_t)b*NN;
    float acc0=0.f, acc1=0.f, acc2=0.f, acc3=0.f;
    for (int n = t; n < NN; n += 256) {
        float dx = a0 - cr[n*3+0], dy = a1 - cr[n*3+1], dz = a2 - cr[n*3+2];
        float d2 = dx*dx + dy*dy + dz*dz;
        float w = wr[n];
        acc0 += w * __expf(-8.0f * d2);                  // std 0.25
        acc1 += w * __expf(-2.0f * d2);                  // std 0.50
        acc2 += w * __expf(-0.8888888888888889f * d2);   // std 0.75
        acc3 += w * __expf(-0.5f * d2);                  // std 1.00
    }
    float s0 = block_reduce(acc0, red);
    float s1 = block_reduce(acc1, red);
    float s2 = block_reduce(acc2, red);
    float s3 = block_reduce(acc3, red);
    if (t == 0) {
        const float TWO_PI = 6.283185307179586f;
        float* ph = ws + WS_PHI + (size_t)bp*NBASIS;
        ph[0] = logf(s0 * powf(TWO_PI*0.0625f, -1.5f) + 1e-4f);
        ph[1] = logf(s1 * powf(TWO_PI*0.25f,   -1.5f) + 1e-4f);
        ph[2] = logf(s2 * powf(TWO_PI*0.5625f, -1.5f) + 1e-4f);
        ph[3] = logf(s3 * powf(TWO_PI*1.0f,    -1.5f) + 1e-4f);
    }
}

// ---------------- FieldEmbedding gate: silu(phi @ Wf^T) * fourier(anchor) ----------------
__global__ __launch_bounds__(256) void k_gate(const float* __restrict__ ffw,
                                              const float* __restrict__ fcw,
                                              float* __restrict__ ws) {
    int bp = blockIdx.x, t = threadIdx.x;
    const float* ph = ws + WS_PHI + (size_t)bp*NBASIS;
    float p0 = ph[0], p1 = ph[1], p2 = ph[2], p3 = ph[3];
    const float* an = ws + WS_ANCH + (size_t)bp*3;
    float a0 = an[0], a1 = an[1], a2 = an[2];
    float* out = ws + WS_H + (size_t)bp*WW;
    for (int w = t; w < WW; w += 256) {
        const float* fw = ffw + w*4;
        float fe = p0*fw[0] + p1*fw[1] + p2*fw[2] + p3*fw[3];
        int ci = w & 511;
        const float* cw = fcw + ci*3;
        float c  = a0*cw[0] + a1*cw[1] + a2*cw[2];
        float ce = (w < 512) ? cosf(c) : sinf(c);
        out[w] = fast_silu(fe) * ce;
    }
}

// ---------------- pairwise anchor distances + std-normalization ----------------
__global__ __launch_bounds__(256) void k_dist(float* __restrict__ ws) {
    int b = blockIdx.y, i = blockIdx.x, t = threadIdx.x;
    const float* A = ws + WS_ANCH + (size_t)b*PP*3;
    float x0 = A[i*3+0], x1 = A[i*3+1], x2 = A[i*3+2];
    float* D = ws + WS_DIST + (size_t)b*PP*PP + (size_t)i*PP;
    for (int j = t; j < PP; j += 256) {
        float d0 = x0 - A[j*3+0] - 1e-5f;
        float d1 = x1 - A[j*3+1] - 1e-5f;
        float d2 = x2 - A[j*3+2] - 1e-5f;
        D[j] = sqrtf(d0*d0 + d1*d1 + d2*d2);
    }
}

__global__ __launch_bounds__(256) void k_dstat(float* __restrict__ ws) {
    __shared__ float red[256];
    int b = blockIdx.x, t = threadIdx.x;
    const float* D = ws + WS_DIST + (size_t)b*PP*PP;
    float s = 0.f, ss = 0.f;
    for (int idx = t; idx < PP*PP; idx += 256) {
        float d = D[idx];
        s += d; ss += d*d;
    }
    float ts  = block_reduce(s,  red);
    float tss = block_reduce(ss, red);
    if (t == 0) { ws[ST_DST + b*2 + 0] = ts; ws[ST_DST + b*2 + 1] = tss; }
}

__global__ __launch_bounds__(256) void k_dscale(float* __restrict__ ws) {
    int idx = blockIdx.x * 256 + threadIdx.x;
    int b = idx / (PP*PP);
    float inv  = 1.0f / (float)(PP*PP);
    float mean = ws[ST_DST + b*2 + 0] * inv;
    float var  = ws[ST_DST + b*2 + 1] * inv - mean*mean;
    ws[WS_DIST + idx] *= rsqrtf(fmaxf(var, 1e-30f));
}

// ---------------- bf16 WMMA GEMM (double-buffered LDS, prefetch) ----------------
// out[M,N] = act(X[M,K] @ Wt[N,K]^T + bias)
// 8 waves; workgroup tile 64x128, wave tile 32x32 (2x2 of 16x16x32 WMMA).
__global__ __launch_bounds__(256) void k_gemm(const float* __restrict__ X,
                                              const float* __restrict__ Wt,
                                              const float* __restrict__ bias,
                                              float* __restrict__ out,
                                              int M, int N, int K, int act) {
    __shared__ __align__(16) __bf16 sA[2][64][40];
    __shared__ __align__(16) __bf16 sB[2][128][40];
    int tid = threadIdx.x;
    int m0 = blockIdx.x * 64;
    int n0 = blockIdx.y * 128;
    int wid  = tid >> 5;
    int lane = tid & 31;
    int wm = wid >> 2, wn = wid & 3;
    int ml = lane & 15, kh = lane >> 4;

    floatx8 acc[2][2];
#pragma unroll
    for (int mi = 0; mi < 2; ++mi)
#pragma unroll
        for (int ni = 0; ni < 2; ++ni)
#pragma unroll
            for (int j = 0; j < 8; ++j) acc[mi][ni][j] = 0.f;

    int arow = tid >> 2, ac0 = (tid & 3) * 8;   // 64 rows x 32 cols, 8 per thread
    int brow = tid >> 1, bc0 = (tid & 1) * 16;  // 128 rows x 32 cols, 16 per thread
    const float* apBase = X  + (size_t)(m0 + arow) * K + ac0;
    const float* bpBase = Wt + (size_t)(n0 + brow) * K + bc0;

    float aR[8];
    float bR[16];
    // prologue: stage K-tile 0 into buffer 0
    {
        float4 a0 = *(const float4*)apBase;
        float4 a1 = *(const float4*)(apBase + 4);
        aR[0]=a0.x; aR[1]=a0.y; aR[2]=a0.z; aR[3]=a0.w;
        aR[4]=a1.x; aR[5]=a1.y; aR[6]=a1.z; aR[7]=a1.w;
        float4 b0 = *(const float4*)bpBase;
        float4 b1 = *(const float4*)(bpBase + 4);
        float4 b2 = *(const float4*)(bpBase + 8);
        float4 b3 = *(const float4*)(bpBase + 12);
        bR[0]=b0.x; bR[1]=b0.y; bR[2]=b0.z; bR[3]=b0.w;
        bR[4]=b1.x; bR[5]=b1.y; bR[6]=b1.z; bR[7]=b1.w;
        bR[8]=b2.x; bR[9]=b2.y; bR[10]=b2.z; bR[11]=b2.w;
        bR[12]=b3.x; bR[13]=b3.y; bR[14]=b3.z; bR[15]=b3.w;
#pragma unroll
        for (int e = 0; e < 8; ++e)  sA[0][arow][ac0 + e] = (__bf16)aR[e];
#pragma unroll
        for (int e = 0; e < 16; ++e) sB[0][brow][bc0 + e] = (__bf16)bR[e];
    }
    __syncthreads();

    int nk = K >> 5;
    for (int kt = 0; kt < nk; ++kt) {
        int cur = kt & 1, nxt = cur ^ 1;
        bool have_next = (kt + 1) < nk;
        if (have_next) {
            const float* ap = apBase + (size_t)(kt + 1) * 32;
            float4 a0 = *(const float4*)ap;
            float4 a1 = *(const float4*)(ap + 4);
            aR[0]=a0.x; aR[1]=a0.y; aR[2]=a0.z; aR[3]=a0.w;
            aR[4]=a1.x; aR[5]=a1.y; aR[6]=a1.z; aR[7]=a1.w;
            const float* bp = bpBase + (size_t)(kt + 1) * 32;
            float4 b0 = *(const float4*)bp;
            float4 b1 = *(const float4*)(bp + 4);
            float4 b2 = *(const float4*)(bp + 8);
            float4 b3 = *(const float4*)(bp + 12);
            bR[0]=b0.x; bR[1]=b0.y; bR[2]=b0.z; bR[3]=b0.w;
            bR[4]=b1.x; bR[5]=b1.y; bR[6]=b1.z; bR[7]=b1.w;
            bR[8]=b2.x; bR[9]=b2.y; bR[10]=b2.z; bR[11]=b2.w;
            bR[12]=b3.x; bR[13]=b3.y; bR[14]=b3.z; bR[15]=b3.w;
            if (kt + 2 < nk) {   // pull the slab after that toward L2/WGP$
                __builtin_prefetch(apBase + (size_t)(kt + 2) * 32, 0, 1);
                __builtin_prefetch(bpBase + (size_t)(kt + 2) * 32, 0, 1);
            }
        }

        FragAB fa[2], fb[2];
#pragma unroll
        for (int mi = 0; mi < 2; ++mi) {
            const __bf16* pa = &sA[cur][wm*32 + mi*16 + ml][kh*16];
            fa[mi].q[0] = *(const uint4*)pa;
            fa[mi].q[1] = *(const uint4*)(pa + 8);
        }
#pragma unroll
        for (int ni = 0; ni < 2; ++ni) {
            const __bf16* pb = &sB[cur][wn*32 + ni*16 + ml][kh*16];
            fb[ni].q[0] = *(const uint4*)pb;
            fb[ni].q[1] = *(const uint4*)(pb + 8);
        }
#pragma unroll
        for (int mi = 0; mi < 2; ++mi)
#pragma unroll
            for (int ni = 0; ni < 2; ++ni)
                acc[mi][ni] = wmma_bf16(fa[mi].v, fb[ni].v, acc[mi][ni]);

        if (have_next) {
#pragma unroll
            for (int e = 0; e < 8; ++e)  sA[nxt][arow][ac0 + e] = (__bf16)aR[e];
#pragma unroll
            for (int e = 0; e < 16; ++e) sB[nxt][brow][bc0 + e] = (__bf16)bR[e];
        }
        __syncthreads();
    }

#pragma unroll
    for (int mi = 0; mi < 2; ++mi)
#pragma unroll
        for (int ni = 0; ni < 2; ++ni) {
            int n = n0 + wn*32 + ni*16 + ml;
            float bs = bias[n];
#pragma unroll
            for (int j = 0; j < 8; ++j) {
                int m = m0 + wm*32 + mi*16 + j + (kh << 3);
                float v = acc[mi][ni][j] + bs;
                if (act == 1) v = fast_silu(v);
                out[(size_t)m * N + n] = v;
            }
        }
}

// ---------------- fused attention: scores (WMMA) - dist bias, softmax, P@V (WMMA) ----------------
__global__ __launch_bounds__(32) void k_attn(const float* __restrict__ Q,
                                             const float* __restrict__ Km,
                                             const float* __restrict__ Vm,
                                             const float* __restrict__ D,
                                             float* __restrict__ O) {
    __shared__ float sc[16][521];
    __shared__ float rinv[16];
    int p0 = blockIdx.x * 16, h = blockIdx.y, b = blockIdx.z;
    int lane = threadIdx.x;
    int ml = lane & 15, kh = lane >> 4;
    const float scale = 0.17677669529663687f;  // 1/sqrt(HD=32)

    // Q fragment (A operand, 16 rows x K=32), pre-scaled
    FragAB fq;
    {
        const float* qp = Q + ((size_t)(b*PP + p0 + ml)) * EE + h*HD_;
#pragma unroll
        for (int e = 0; e < 8; ++e) fq.h[e]     = (__bf16)(qp[kh*16 + e] * scale);
#pragma unroll
        for (int e = 0; e < 8; ++e) fq.h[8 + e] = (__bf16)(qp[kh*16 + 8 + e] * scale);
    }

    // scores: 32 tiles of 16 keys
    const float* dp = D + (size_t)b*PP*PP;
    for (int kt = 0; kt < PP/16; ++kt) {
        int kb0 = kt * 16;
        FragAB fk;  // B operand: [K=32 dims x N=16 keys]
        const float* kp = Km + ((size_t)(b*PP + kb0 + ml)) * EE + h*HD_ + kh*16;
#pragma unroll
        for (int e = 0; e < 16; ++e) fk.h[e] = (__bf16)kp[e];
        floatx8 accS;
#pragma unroll
        for (int j = 0; j < 8; ++j) accS[j] = 0.f;
        accS = wmma_bf16(fq.v, fk.v, accS);
#pragma unroll
        for (int j = 0; j < 8; ++j) {
            int mr = j + (kh << 3);
            sc[mr][kb0 + ml] = accS[j] - dp[(size_t)(p0 + mr)*PP + kb0 + ml];
        }
    }
    __syncthreads();

    // row-wise softmax over 512 keys
    for (int r = 0; r < 16; ++r) {
        float mx = -3.4e38f;
        for (int c = lane; c < PP; c += 32) mx = fmaxf(mx, sc[r][c]);
#pragma unroll
        for (int off = 16; off; off >>= 1) mx = fmaxf(mx, __shfl_xor(mx, off, 32));
        float sm = 0.f;
        for (int c = lane; c < PP; c += 32) {
            float e = __expf(sc[r][c] - mx);
            sc[r][c] = e;
            sm += e;
        }
#pragma unroll
        for (int off = 16; off; off >>= 1) sm += __shfl_xor(sm, off, 32);
        if (lane == 0) rinv[r] = fast_rcp(sm);
    }
    __syncthreads();

    // P @ V : 16 K-steps of 32 keys; output 16 rows x 32 dims (two 16x16 tiles)
    floatx8 acc0, acc1;
#pragma unroll
    for (int j = 0; j < 8; ++j) { acc0[j] = 0.f; acc1[j] = 0.f; }
    for (int kt = 0; kt < PP/32; ++kt) {
        int kb0 = kt * 32;
        FragAB fpr;  // probabilities A operand
#pragma unroll
        for (int e = 0; e < 8; ++e) fpr.h[e]     = (__bf16)sc[ml][kb0 + kh*16 + e];
#pragma unroll
        for (int e = 0; e < 8; ++e) fpr.h[8 + e] = (__bf16)sc[ml][kb0 + kh*16 + 8 + e];
        FragAB fv0, fv1;  // V B operand: [32 keys x 16 dims]
        const float* vp = Vm + ((size_t)(b*PP + kb0 + kh*16)) * EE + h*HD_;
#pragma unroll
        for (int e = 0; e < 16; ++e) {
            const float* vpe = vp + (size_t)e * EE;
            fv0.h[e] = (__bf16)vpe[ml];
            fv1.h[e] = (__bf16)vpe[16 + ml];
        }
        acc0 = wmma_bf16(fpr.v, fv0.v, acc0);
        acc1 = wmma_bf16(fpr.v, fv1.v, acc1);
    }
    float* op = O + ((size_t)(b*PP + p0)) * EE + h*HD_;
#pragma unroll
    for (int j = 0; j < 8; ++j) {
        int mr = j + (kh << 3);
        float ri = rinv[mr];
        op[(size_t)mr*EE + ml]      = acc0[j] * ri;
        op[(size_t)mr*EE + 16 + ml] = acc1[j] * ri;
    }
}

// ---------------- residual add + layernorm ----------------
__global__ __launch_bounds__(256) void k_addln(const float* __restrict__ Xin,
                                               const float* __restrict__ Rsd,
                                               const float* __restrict__ g,
                                               const float* __restrict__ bta,
                                               float* __restrict__ Xout) {
    __shared__ float red[256];
    int row = blockIdx.x, t = threadIdx.x;
    float v = Xin[(size_t)row*EE + t] + Rsd[(size_t)row*EE + t];
    float mean = block_reduce(v, red) * (1.0f / EE);
    float d = v - mean;
    float var = block_reduce(d*d, red) * (1.0f / EE);
    Xout[(size_t)row*EE + t] = d * rsqrtf(var + 1e-5f) * g[t] + bta[t];
}

// ---------------- FieldProjection head: sum over P, LN, project, combine ----------------
__global__ __launch_bounds__(256) void k_final(const float* __restrict__ Y,
                                               const float* __restrict__ ws,
                                               const float* __restrict__ fpng,
                                               const float* __restrict__ fpnb,
                                               const float* __restrict__ fppw,
                                               const float* __restrict__ fppb,
                                               float* __restrict__ out) {
    __shared__ float red[256];
    int b = blockIdx.x, t = threadIdx.x;
    float s = 0.f;
    for (int p = 0; p < PP; ++p) s += Y[((size_t)(b*PP + p)) * EE + t];
    float mean = block_reduce(s, red) * (1.f / EE);
    float d = s - mean;
    float var = block_reduce(d*d, red) * (1.f / EE);
    float z = d * rsqrtf(var + 1e-5f) * fpng[t] + fpnb[t];
    float d0 = block_reduce(z * fppw[t],       red);
    float d1 = block_reduce(z * fppw[EE + t],  red);
    if (t == 0) {
        float y0 = d0 + fppb[0], y1 = d1 + fppb[1];
        float scl = (y0 > 20.f) ? y0 : log1pf(expf(y0));                          // softplus
        float sp  = (-2.f*y1 > 20.f) ? (-2.f*y1) : log1pf(expf(-2.f*y1));
        float lc  = y1 + sp - LN2_;                                               // log_cosh
        out[b] = scl * ws[ST_ELDA + b] - lc;
    }
}

// ---------------- host orchestration ----------------
extern "C" void kernel_launch(void* const* d_in, const int* in_sizes, int n_in,
                              void* d_out, int out_size, void* d_ws, size_t ws_size,
                              hipStream_t stream) {
    const float* rho     = (const float*)d_in[0];
    const float* coords  = (const float*)d_in[1];
    const float* weights = (const float*)d_in[2];
    const float* grid    = (const float*)d_in[3];
    const float* ffw     = (const float*)d_in[4];   // fe_field_w [W,4]
    const float* fcw     = (const float*)d_in[5];   // fe_coord_w [W/2,3]
    const float* fepw    = (const float*)d_in[6];   // fe_proj_w [E,W]
    const float* fepb    = (const float*)d_in[7];
    const float* wq      = (const float*)d_in[8];
    const float* wk      = (const float*)d_in[9];
    const float* wv      = (const float*)d_in[10];
    const float* wo      = (const float*)d_in[11];
    const float* bq      = (const float*)d_in[12];
    const float* bk      = (const float*)d_in[13];
    const float* bv      = (const float*)d_in[14];
    const float* bo      = (const float*)d_in[15];
    const float* mw1     = (const float*)d_in[16];
    const float* mb1     = (const float*)d_in[17];
    const float* mw2     = (const float*)d_in[18];
    const float* mb2     = (const float*)d_in[19];
    const float* ng1     = (const float*)d_in[20];
    const float* nb1     = (const float*)d_in[21];
    const float* ng2     = (const float*)d_in[22];
    const float* nb2     = (const float*)d_in[23];
    const float* fpw1    = (const float*)d_in[24];
    const float* fpb1    = (const float*)d_in[25];
    const float* fpw2    = (const float*)d_in[26];
    const float* fpb2    = (const float*)d_in[27];
    const float* fpng    = (const float*)d_in[28];
    const float* fpnb    = (const float*)d_in[29];
    const float* fppw    = (const float*)d_in[30];
    const float* fppb    = (const float*)d_in[31];
    float* ws  = (float*)d_ws;
    float* out = (float*)d_out;
    const int M = BB * PP;  // 2048

    k_stats1<<<BB, 256, 0, stream>>>(rho, coords, weights, ws);
    k_cov<<<BB, 256, 0, stream>>>(coords, ws);
    k_eigen<<<1, 32, 0, stream>>>(ws);
    k_anchors<<<(BB*PP + 255)/256, 256, 0, stream>>>(grid, ws);
    k_coordsr<<<(BB*NN + 255)/256, 256, 0, stream>>>(coords, ws);
    k_gauss<<<BB*PP, 256, 0, stream>>>(ws);
    k_gate<<<BB*PP, 256, 0, stream>>>(ffw, fcw, ws);
    k_gemm<<<dim3(M/64, EE/128), 256, 0, stream>>>(ws+WS_H, fepw, fepb, ws+WS_X, M, EE, WW, 0);
    k_dist<<<dim3(PP, BB), 256, 0, stream>>>(ws);
    k_dstat<<<BB, 256, 0, stream>>>(ws);
    k_dscale<<<(BB*PP*PP)/256, 256, 0, stream>>>(ws);

    for (int i = 0; i < NBLOCKS_; ++i) {
        k_gemm<<<dim3(M/64, EE/128), 256, 0, stream>>>(ws+WS_X, wq + (size_t)i*EE*EE, bq + i*EE, ws+WS_Q, M, EE, EE, 0);
        k_gemm<<<dim3(M/64, EE/128), 256, 0, stream>>>(ws+WS_X, wk + (size_t)i*EE*EE, bk + i*EE, ws+WS_K, M, EE, EE, 0);
        k_gemm<<<dim3(M/64, EE/128), 256, 0, stream>>>(ws+WS_X, wv + (size_t)i*EE*EE, bv + i*EE, ws+WS_V, M, EE, EE, 0);
        k_attn<<<dim3(PP/16, NHEADS_, BB), 32, 0, stream>>>(ws+WS_Q, ws+WS_K, ws+WS_V, ws+WS_DIST, ws+WS_O);
        k_gemm<<<dim3(M/64, EE/128), 256, 0, stream>>>(ws+WS_O, wo + (size_t)i*EE*EE, bo + i*EE, ws+WS_O2, M, EE, EE, 0);
        k_addln<<<M, 256, 0, stream>>>(ws+WS_X, ws+WS_O2, ng1 + i*EE, nb1 + i*EE, ws+WS_X);
        k_gemm<<<dim3(M/64, WW/128), 256, 0, stream>>>(ws+WS_X, mw1 + (size_t)i*WW*EE, mb1 + i*WW, ws+WS_H, M, WW, EE, 1);
        k_gemm<<<dim3(M/64, EE/128), 256, 0, stream>>>(ws+WS_H, mw2 + (size_t)i*EE*WW, mb2 + i*EE, ws+WS_O2, M, EE, WW, 0);
        k_addln<<<M, 256, 0, stream>>>(ws+WS_X, ws+WS_O2, ng2 + i*EE, nb2 + i*EE, ws+WS_X);
    }

    k_gemm<<<dim3(M/64, WW/128), 256, 0, stream>>>(ws+WS_X, fpw1, fpb1, ws+WS_H, M, WW, EE, 1);
    k_gemm<<<dim3(M/64, EE/128), 256, 0, stream>>>(ws+WS_H, fpw2, fpb2, ws+WS_O2, M, EE, WW, 0);
    k_final<<<BB, 256, 0, stream>>>(ws+WS_O2, ws, fpng, fpnb, fppw, fppb, out);

    (void)in_sizes; (void)n_in; (void)out_size; (void)ws_size;
}
